// LSTM_RNN_43233140801797
// MI455X (gfx1250) — matrix operand order repaired and checked by reading, not compiled, not run on hardware
//
#include <hip/hip_runtime.h>

// ---------------------------------------------------------------------------
// 2-layer LSTM (B=8192, T=2048, H=51) for gfx1250.
// bf16 WMMA 16x16x32, f32 accumulation, accumulators start from inline-0 C:
// x_t, input weight and both biases are folded into padded K slots
//   K=51 : x_t        (W1[g][51] = wih1[g],  W2[g][51] = 0)
//   K=52 : constant 1 (W1[g][52] = bias1[g], W2[g][52] = bias2[g])
// Gate rows unit-interleaved (r = 4u + {i,f,g,o}) -> nonlinearity reads all
// four gates of a unit with one ds_load_b128. LDS strides padded for 64-bank
// conflict freedom. c-state in registers. 8 waves/block (2 per SIMD) for
// latency hiding in the serial recurrence; B-fragment loads software-
// pipelined one tile ahead to decouple them from the shared DScnt waits.
// ---------------------------------------------------------------------------

typedef __attribute__((ext_vector_type(16))) __bf16 v16bf;
typedef __attribute__((ext_vector_type(8)))  __bf16 v8bf;
typedef __attribute__((ext_vector_type(8)))  float  v8f;

namespace {
constexpr int Hh    = 51;            // hidden size
constexpr int GP    = 208;           // padded gate dim = 13 * 16
constexpr int KP    = 64;            // padded K dim     = 2 * 32
constexpr int NT    = 13;            // N tiles of 16
constexpr int Bb    = 8192;
constexpr int Tt    = 2048;
constexpr int WAVES = 8;             // 256 threads: 2 waves per SIMD32
constexpr int ROWS  = 16;            // batch rows per wave

constexpr int KX    = 51;            // K slot carrying x_t (layer 1 only)
constexpr int KONE  = 52;            // K slot carrying constant 1 (bias row)

// LDS strides padded for 64-bank conflict freedom (keep 16B/32B alignment)
constexpr int WS = 80;               // weight row stride (bf16): 160B
constexpr int HS = 72;               // h-state row stride (bf16): 144B
constexpr int GS = 212;              // gate buffer row stride (f32): 848B

constexpr int WBE = GP * WS;         // bf16 elements per padded weight matrix

constexpr int PW_BYTES   = 16*GS*4 /*gates*/ + 2*16*HS*2 /*h1,h2*/;   // 18176
constexpr int SMEM_BYTES = 3*WBE*2 + KP*4 + WAVES*PW_BYTES;           // 245504 < 320KB
}

// fast gate nonlinearities via hardware v_exp_f32 / v_rcp_f32
__device__ __forceinline__ float sigf(float x) {
  return __frcp_rn(1.0f + __expf(-x));
}
__device__ __forceinline__ float tanhf_fast(float x) {
  return 1.0f - 2.0f * __frcp_rn(__expf(2.0f * x) + 1.0f);
}

// A-matrix fragment, contiguous K order (matches loadB):
//   lanes 0-15 : M=lane,    elems 0-15 -> K=kbase..+15
//   lanes16-31 : M=lane-16, elems 0-15 -> K=kbase+16..+31
__device__ __forceinline__ v16bf loadA(const __bf16* row, int kbase, int hi) {
  v8bf lo = *(const v8bf*)(row + kbase + hi * 16);
  v8bf hh = *(const v8bf*)(row + kbase + hi * 16 + 8);
  v16bf r;
#pragma unroll
  for (int e = 0; e < 8; ++e) { r[e] = lo[e]; r[8 + e] = hh[e]; }
  return r;
}

// B-matrix fragment: lane = column N, 16 contiguous K values (32B load)
__device__ __forceinline__ v16bf loadB(const __bf16* W, int g, int kbase, int hi) {
  return *(const v16bf*)(W + g * WS + kbase + hi * 16);
}

// ---------------------------------------------------------------------------
// Prep: bf16 B-matrix layout Wb[n][k], unit-interleaved gate rows:
// padded row n -> unit u = n>>2, gate p = n&3, original row orig = p*H + u
// (PyTorch order i,f,g,o). x-weight and biases folded into K slots 51/52.
// ---------------------------------------------------------------------------
__global__ void prep_weights(const float* __restrict__ Wih1, const float* __restrict__ Whh1,
                             const float* __restrict__ bih1, const float* __restrict__ bhh1,
                             const float* __restrict__ Wih2, const float* __restrict__ Whh2,
                             const float* __restrict__ bih2, const float* __restrict__ bhh2,
                             const float* __restrict__ Wlin,
                             __bf16* __restrict__ Wb1, __bf16* __restrict__ Wb2,
                             __bf16* __restrict__ Wb3, float* __restrict__ wlinp)
{
  const int idx    = blockIdx.x * blockDim.x + threadIdx.x;
  const int stride = gridDim.x * blockDim.x;
  for (int i = idx; i < GP * WS; i += stride) {
    const int n = i / WS, k = i % WS;
    const int u = n >> 2, p = n & 3;
    const int orig = p * Hh + u;
    float w1 = 0.0f, w2 = 0.0f, w3 = 0.0f;
    if (u < Hh) {
      if (k < Hh) {
        w1 = Whh1[orig * Hh + k];
        w2 = Wih2[orig * Hh + k];
        w3 = Whh2[orig * Hh + k];
      } else if (k == KX) {
        w1 = Wih1[orig];                       // x_t weight (layer 1 only)
      } else if (k == KONE) {
        w1 = bih1[orig] + bhh1[orig];          // layer-1 bias
        w2 = bih2[orig] + bhh2[orig];          // layer-2 bias
      }
    }
    Wb1[i] = (__bf16)w1;
    Wb2[i] = (__bf16)w2;
    Wb3[i] = (__bf16)w3;
  }
  for (int k = idx; k < KP; k += stride) wlinp[k] = (k < Hh) ? Wlin[k] : 0.0f;
}

// ---------------------------------------------------------------------------
// Main recurrent kernel.
// ---------------------------------------------------------------------------
__global__ void __launch_bounds__(WAVES * 32, 1)
lstm2_wmma(const float* __restrict__ x,
           const __bf16* __restrict__ Wb1, const __bf16* __restrict__ Wb2,
           const __bf16* __restrict__ Wb3,
           const float* __restrict__ wlin, const float* __restrict__ blin,
           float* __restrict__ out)
{
  extern __shared__ char smem[];
  __bf16* sW1   = (__bf16*)smem;           // [GP][WS] bf16
  __bf16* sW2   = sW1 + WBE;
  __bf16* sW3   = sW2 + WBE;
  float*  swlin = (float*)(sW3 + WBE);     // [KP]

  const int tid   = threadIdx.x;
  const int lane  = tid & 31;
  const int wave  = tid >> 5;
  const int laneN = lane & 15;
  const int hi    = lane >> 4;
  const int r0    = blockIdx.x * (WAVES * ROWS) + wave * ROWS;

  char*   pw  = (char*)(swlin + KP) + wave * PW_BYTES;   // wave-private region
  float*  sG  = (float*)pw;                 // [16][GS] f32 gate buffer
  __bf16* sh1 = (__bf16*)(sG + 16 * GS);    // [16][HS] bf16 h1 state (+x, +1)
  __bf16* sh2 = sh1 + 16 * HS;              // [16][HS] bf16 h2 state

  // ---- stage weights into LDS (reused 2048x) ----
  for (int i = tid; i < WBE; i += blockDim.x) { sW1[i] = Wb1[i]; sW2[i] = Wb2[i]; sW3[i] = Wb3[i]; }
  for (int i = tid; i < KP; i += blockDim.x) { swlin[i] = wlin[i]; }
  for (int i = lane; i < ROWS * HS; i += 32) { sh1[i] = (__bf16)0.0f; sh2[i] = (__bf16)0.0f; }
  __builtin_amdgcn_wave_barrier();
  if (lane < ROWS) sh1[lane * HS + KONE] = (__bf16)1.0f;   // bias-row constant
  __syncthreads();

  float c1r[26], c2r[26];
#pragma unroll
  for (int q = 0; q < 26; ++q) { c1r[q] = 0.0f; c2r[q] = 0.0f; }

  const float blv = blin[0];
  const int   ub  = hi * 26;                // hidden-unit sub-range per half-wave
  const __bf16* arow1 = sh1 + laneN * HS;
  const __bf16* arow2 = sh2 + laneN * HS;

  for (int t = 0; t < Tt; ++t) {
    // x_t for the 16 rows of this wave goes straight into the K=51 slot
    if (lane < ROWS) {
      const size_t xo = (size_t)(r0 + lane) * Tt + t;
      sh1[lane * HS + KX] = (__bf16)x[xo];
      if (t + 32 < Tt) __builtin_prefetch(x + xo + 32, 0, 1);
    }
    __builtin_amdgcn_wave_barrier();

    // ========== layer 1: G = [h1|x|1] @ [Whh1|wih1|b1]^T  (C starts at 0) ==========
    {
      v16bf a0 = loadA(arow1, 0, hi);
      v16bf a1 = loadA(arow1, 32, hi);
      // software pipeline: B fragments loaded one tile ahead
      v16bf nb0 = loadB(sW1, laneN, 0, hi);
      v16bf nb1 = loadB(sW1, laneN, 32, hi);
#pragma unroll
      for (int j = 0; j < NT; ++j) {
        const int g = j * 16 + laneN;
        v16bf b0 = nb0, b1 = nb1;
        if (j + 1 < NT) {
          nb0 = loadB(sW1, g + 16, 0, hi);
          nb1 = loadB(sW1, g + 16, 32, hi);
        }
        v8f acc = (v8f)(0.0f);
        acc = __builtin_amdgcn_wmma_f32_16x16x32_bf16(false, a0, false, b0, (short)0, acc, false, false);
        acc = __builtin_amdgcn_wmma_f32_16x16x32_bf16(false, a1, false, b1, (short)0, acc, false, false);
#pragma unroll
        for (int v = 0; v < 8; ++v) sG[(v + 8 * hi) * GS + g] = acc[v];
      }
    }
    __builtin_amdgcn_wave_barrier();

    // ---- layer-1 gates -> c1, h1 : one float4 = {i,f,g,o} per unit ----
    {
      const float* Gr   = sG + laneN * GS;
      __bf16*      hrow = sh1 + laneN * HS;
#pragma unroll
      for (int q = 0; q < 26; ++q) {
        const int u = ub + q;
        if (u < Hh) {
          const float4 gv = *(const float4*)(Gr + 4 * u);
          const float cn = sigf(gv.y) * c1r[q] + sigf(gv.x) * tanhf_fast(gv.z);
          c1r[q] = cn;
          hrow[u] = (__bf16)(sigf(gv.w) * tanhf_fast(cn));
        }
      }
    }
    __builtin_amdgcn_wave_barrier();

    // ========== layer 2: G = [h1|0|1] @ [Wih2|0|b2]^T + h2 @ Whh2^T ==========
    {
      v16bf a0 = loadA(arow1, 0, hi);
      v16bf a1 = loadA(arow1, 32, hi);
      v16bf a2 = loadA(arow2, 0, hi);
      v16bf a3 = loadA(arow2, 32, hi);
      // pipeline the W2 pair one tile ahead; W3 pair just-in-time (covered by
      // the two preceding WMMAs) to stay under the 256-VGPR cliff
      v16bf nb0 = loadB(sW2, laneN, 0, hi);
      v16bf nb1 = loadB(sW2, laneN, 32, hi);
#pragma unroll
      for (int j = 0; j < NT; ++j) {
        const int g = j * 16 + laneN;
        v16bf b0 = nb0, b1 = nb1;
        if (j + 1 < NT) {
          nb0 = loadB(sW2, g + 16, 0, hi);
          nb1 = loadB(sW2, g + 16, 32, hi);
        }
        v16bf b2 = loadB(sW3, g, 0, hi);
        v16bf b3 = loadB(sW3, g, 32, hi);
        v8f acc = (v8f)(0.0f);
        acc = __builtin_amdgcn_wmma_f32_16x16x32_bf16(false, a0, false, b0, (short)0, acc, false, false);
        acc = __builtin_amdgcn_wmma_f32_16x16x32_bf16(false, a1, false, b1, (short)0, acc, false, false);
        acc = __builtin_amdgcn_wmma_f32_16x16x32_bf16(false, a2, false, b2, (short)0, acc, false, false);
        acc = __builtin_amdgcn_wmma_f32_16x16x32_bf16(false, a3, false, b3, (short)0, acc, false, false);
#pragma unroll
        for (int v = 0; v < 8; ++v) sG[(v + 8 * hi) * GS + g] = acc[v];
      }
    }
    __builtin_amdgcn_wave_barrier();

    // ---- layer-2 gates -> c2, h2 + output projection y = h2 . wlin + blin ----
    float ysum = 0.0f;
    {
      const float* Gr   = sG + laneN * GS;
      __bf16*      hrow = sh2 + laneN * HS;
#pragma unroll
      for (int q = 0; q < 26; ++q) {
        const int u = ub + q;
        if (u < Hh) {
          const float4 gv = *(const float4*)(Gr + 4 * u);
          const float cn = sigf(gv.y) * c2r[q] + sigf(gv.x) * tanhf_fast(gv.z);
          c2r[q] = cn;
          const float hn = sigf(gv.w) * tanhf_fast(cn);
          hrow[u] = (__bf16)hn;
          ysum = __builtin_fmaf(hn, swlin[u], ysum);
        }
      }
    }
    // combine the two half-wave partial dots (lane L += lane L+16)
    const float other = __shfl_down(ysum, 16, 32);
    if (lane < ROWS) out[(size_t)(r0 + laneN) * Tt + t] = ysum + other + blv;
    __builtin_amdgcn_wave_barrier();
  }
}

// ---------------------------------------------------------------------------
extern "C" void kernel_launch(void* const* d_in, const int* in_sizes, int n_in,
                              void* d_out, int out_size, void* d_ws, size_t ws_size,
                              hipStream_t stream) {
  (void)in_sizes; (void)n_in; (void)out_size; (void)ws_size;
  const float* x    = (const float*)d_in[0];
  const float* Wih1 = (const float*)d_in[1];
  const float* Whh1 = (const float*)d_in[2];
  const float* bih1 = (const float*)d_in[3];
  const float* bhh1 = (const float*)d_in[4];
  const float* Wih2 = (const float*)d_in[5];
  const float* Whh2 = (const float*)d_in[6];
  const float* bih2 = (const float*)d_in[7];
  const float* bhh2 = (const float*)d_in[8];
  const float* Wlin = (const float*)d_in[9];
  const float* blin = (const float*)d_in[10];
  float* out = (float*)d_out;

  // workspace layout (bytes)
  char* ws = (char*)d_ws;
  __bf16* Wb1   = (__bf16*)ws;                 ws += (size_t)WBE * 2;
  __bf16* Wb2   = (__bf16*)ws;                 ws += (size_t)WBE * 2;
  __bf16* Wb3   = (__bf16*)ws;                 ws += (size_t)WBE * 2;
  float*  wlinp = (float*)ws;                  ws += (size_t)KP * 4;

  prep_weights<<<64, 256, 0, stream>>>(Wih1, Whh1, bih1, bhh1, Wih2, Whh2, bih2, bhh2,
                                       Wlin, Wb1, Wb2, Wb3, wlinp);

  (void)hipFuncSetAttribute((const void*)lstm2_wmma,
                            hipFuncAttributeMaxDynamicSharedMemorySize, SMEM_BYTES);

  const int blocks = Bb / (WAVES * ROWS);      // 64 blocks of 8 waves
  lstm2_wmma<<<blocks, WAVES * 32, SMEM_BYTES, stream>>>(
      x, Wb1, Wb2, Wb3, wlinp, blin, out);
}
